// GRUBlock_33406255628924
// MI455X (gfx1250) — matrix operand order repaired and checked by reading, not compiled, not run on hardware
//
#include <hip/hip_runtime.h>
#include <hip/hip_bf16.h>

// ---------------- types ----------------
typedef __attribute__((ext_vector_type(16))) __bf16 bf16x16;
typedef __attribute__((ext_vector_type(8)))  float  f32x8;
typedef __attribute__((ext_vector_type(4)))  unsigned int u32x4;

struct B32 { u32x4 lo; u32x4 hi; };   // 32-byte fragment container

static __device__ __forceinline__ unsigned short f32_to_bf16_rne(float f) {
    unsigned int u = __float_as_uint(f);
    u += 0x7FFFu + ((u >> 16) & 1u);
    return (unsigned short)(u >> 16);
}

// gfx1250 async global->LDS copy (ASYNCcnt tracked, no VGPR staging).
// Low 32 bits of a generic pointer to LDS are the raw LDS byte offset.
static __device__ __forceinline__ void async_copy_b128(void* lds_ptr, const void* gptr) {
    unsigned lds_addr = (unsigned)(uintptr_t)lds_ptr;
    unsigned long long ga = (unsigned long long)(uintptr_t)gptr;
    asm volatile("global_load_async_to_lds_b128 %0, %1, off"
                 :: "v"(lds_addr), "v"(ga) : "memory");
}
static __device__ __forceinline__ void wait_async0() {
    asm volatile("s_wait_asynccnt 0x0" ::: "memory");
}

// ---------------- constants ----------------
#define BB   4
#define TT   8192
#define HH   1024
#define RR   1024
#define MM   (BB * TT)        // 32768
#define NCH  64               // scan chunks
#define LCH  (TT / NCH)       // 128 steps per chunk

// ---------------- kernel 0: weight fp32 -> bf16 ----------------
__global__ __launch_bounds__(256)
void k_cvt_w(const float* __restrict__ w_zh, const float* __restrict__ w_out,
             unsigned short* __restrict__ wzh_bf, unsigned short* __restrict__ wout_bf) {
    size_t i = (size_t)blockIdx.x * 256 + threadIdx.x;
    const size_t NZ = (size_t)2 * RR * HH;
    const size_t NO = (size_t)HH * RR;
    if (i < NZ)            wzh_bf[i]       = f32_to_bf16_rne(w_zh[i]);
    else if (i < NZ + NO)  wout_bf[i - NZ] = f32_to_bf16_rne(w_out[i - NZ]);
}

// ---------------- kernel 1: residual causal conv -> bf16 A-matrix ----------------
__global__ __launch_bounds__(256)
void k_conv(const float* __restrict__ x, const float* __restrict__ conv_w,
            const float* __restrict__ conv_b, unsigned short* __restrict__ A1) {
    const int TC = 32;
    int gid   = blockIdx.x * 256 + threadIdx.x;
    int k     = gid & (HH - 1);
    int rest  = gid >> 10;
    int chunk = rest & (TT / TC - 1);
    int b     = rest >> 8;
    if (b >= BB) return;
    float w0 = conv_w[0 * HH + k], w1 = conv_w[1 * HH + k];
    float w2 = conv_w[2 * HH + k], w3 = conv_w[3 * HH + k];
    float bias = conv_b[k];
    int t0 = chunk * TC;
    const float* xb = x + (size_t)b * TT * HH + k;
    float xm3 = (t0 >= 3) ? xb[(size_t)(t0 - 3) * HH] : 0.f;
    float xm2 = (t0 >= 2) ? xb[(size_t)(t0 - 2) * HH] : 0.f;
    float xm1 = (t0 >= 1) ? xb[(size_t)(t0 - 1) * HH] : 0.f;
    for (int t = t0; t < t0 + TC; ++t) {
        float xc = xb[(size_t)t * HH];
        float y  = xc + bias + w0 * xm3 + w1 * xm2 + w2 * xm1 + w3 * xc;
        A1[((size_t)b * TT + t) * HH + k] = f32_to_bf16_rne(y);
        xm3 = xm2; xm2 = xm1; xm1 = xc;
    }
}

// ---------------- WMMA GEMM: C[m,n] = sum_k A_bf16[m,k] * W_bf16[n,k] ----------------
// Block tile 128x128, BLK_K = 64 (two wmma K steps per stage), double-buffered
// LDS fed by global_load_async_to_lds_b128. 8 waves: 4(M) x 2(N), each wave owns
// a 32x64 tile = 2x4 f32 accumulators -> 16 v_wmma per stage per wave.
__global__ __launch_bounds__(256)
void k_gemm_bf16(const unsigned short* __restrict__ A,
                 const unsigned short* __restrict__ Bw,
                 float* __restrict__ C, int M, int N, int K) {
    __shared__ __align__(16) unsigned short As[2][128 * 64];
    __shared__ __align__(16) unsigned short Bs[2][128 * 64];

    const int tid   = threadIdx.x;
    const int lane  = tid & 31;           // wave32
    const int wave  = tid >> 5;
    const int wm    = wave & 3;           // M sub-tile: wm*32
    const int wn    = wave >> 2;          // N sub-tile: wn*64
    const int n0    = blockIdx.x * 128;
    const int m0    = blockIdx.y * 128;
    const int lhalf = lane >> 4;
    const int l15   = lane & 15;

    f32x8 acc[2][4] = {};

    const int S = K >> 6;                 // stages of 64 K

    // ---- stage 0 prologue: async fill buffer 0 ----
    #pragma unroll
    for (int i = 0; i < 4; ++i) {
        int idx = tid + i * 256;          // 0..1023
        int row = idx >> 3;               // 128 rows
        int c8  = idx & 7;                // 8 x 16B per row (64 ushorts)
        async_copy_b128(&As[0][row * 64 + c8 * 8], A  + (size_t)(m0 + row) * K + c8 * 8);
        async_copy_b128(&Bs[0][row * 64 + c8 * 8], Bw + (size_t)(n0 + row) * K + c8 * 8);
    }
    wait_async0();
    __syncthreads();

    for (int s = 0; s < S; ++s) {
        const int p = s & 1;
        // ---- kick off next stage into the other buffer (clamped; last iter is dead) ----
        int knx = (s + 1 < S) ? (s + 1) << 6 : (S - 1) << 6;
        #pragma unroll
        for (int i = 0; i < 4; ++i) {
            int idx = tid + i * 256;
            int row = idx >> 3;
            int c8  = idx & 7;
            async_copy_b128(&As[1 - p][row * 64 + c8 * 8],
                            A  + (size_t)(m0 + row) * K + knx + c8 * 8);
            async_copy_b128(&Bs[1 - p][row * 64 + c8 * 8],
                            Bw + (size_t)(n0 + row) * K + knx + c8 * 8);
        }

        // ---- compute current stage: two 32-deep WMMA K-steps ----
        #pragma unroll
        for (int kk = 0; kk < 64; kk += 32) {
            bf16x16 av[2], bv[4];
            #pragma unroll
            for (int fm = 0; fm < 2; ++fm) {
                int mrow = wm * 32 + fm * 16 + l15;          // A: M = lane%16
                const u32x4* q = (const u32x4*)(&As[p][mrow * 64 + kk]);
                B32 t; t.lo = q[lhalf]; t.hi = q[2 + lhalf]; // K 0-7/16-23 | 8-15/24-31
                av[fm] = __builtin_bit_cast(bf16x16, t);
            }
            #pragma unroll
            for (int fn = 0; fn < 4; ++fn) {
                int nrow = wn * 64 + fn * 16 + l15;          // B: N = lane%16
                const u32x4* q = (const u32x4*)(&Bs[p][nrow * 64 + kk]);
                B32 t; t.lo = q[lhalf]; t.hi = q[2 + lhalf];
                bv[fn] = __builtin_bit_cast(bf16x16, t);
            }
            #pragma unroll
            for (int fm = 0; fm < 2; ++fm)
                #pragma unroll
                for (int fn = 0; fn < 4; ++fn)
                    acc[fm][fn] = __builtin_amdgcn_wmma_f32_16x16x32_bf16(
                        false, av[fm], false, bv[fn], (short)0, acc[fm][fn], false, false);
        }

        // ---- all async writes for next stage landed, all reads of p done ----
        wait_async0();
        __syncthreads();
    }

    // ---- epilogue: D layout — VGPR v holds M = v + (lane>=16)*8, N = lane%16 ----
    #pragma unroll
    for (int fm = 0; fm < 2; ++fm) {
        int mbase = m0 + wm * 32 + fm * 16 + lhalf * 8;
        #pragma unroll
        for (int fn = 0; fn < 4; ++fn) {
            int n = n0 + wn * 64 + fn * 16 + l15;
            #pragma unroll
            for (int v = 0; v < 8; ++v)
                C[(size_t)(mbase + v) * N + n] = acc[fm][fn][v];
        }
    }
}

// ---------------- scan pass 1: per-chunk affine carry (A,B) ----------------
__global__ __launch_bounds__(256)
void k_scan1(const float* __restrict__ zh, float* __restrict__ cA, float* __restrict__ cB) {
    int gid  = blockIdx.x * 256 + threadIdx.x;   // [b][chunk][r]
    int r    = gid & (RR - 1);
    int rest = gid >> 10;
    int c    = rest & (NCH - 1);
    int b    = rest >> 6;
    if (b >= BB) return;
    const float* base = zh + ((size_t)b * TT + (size_t)c * LCH) * (2 * RR);
    float Ac = 1.f, Bc = 0.f;
    for (int i = 0; i < LCH; ++i) {
        float zl = base[(size_t)i * (2 * RR) + r];
        float hl = base[(size_t)i * (2 * RR) + RR + r];
        float z  = 1.f / (1.f + __expf(-zl));
        float a  = 1.f - z;
        Ac = a * Ac;
        Bc = a * Bc + z * hl;
    }
    cA[gid] = Ac; cB[gid] = Bc;
}

// ---------------- scan pass 2: inter-chunk prefix ----------------
__global__ __launch_bounds__(256)
void k_scan2(const float* __restrict__ cA, const float* __restrict__ cB,
             float* __restrict__ hinit) {
    int gid = blockIdx.x * 256 + threadIdx.x;    // b*RR + r
    if (gid >= BB * RR) return;
    int r = gid & (RR - 1); int b = gid >> 10;
    float h = 0.f;
    for (int c = 0; c < NCH; ++c) {
        size_t idx = ((size_t)b * NCH + c) * RR + r;
        hinit[idx] = h;
        h = cA[idx] * h + cB[idx];
    }
}

// ---------------- scan pass 3: replay with correct h0, emit bf16 h_o ----------------
__global__ __launch_bounds__(256)
void k_scan3(const float* __restrict__ zh, const float* __restrict__ hinit,
             unsigned short* __restrict__ ho) {
    int gid  = blockIdx.x * 256 + threadIdx.x;   // [b][chunk][r]
    int r    = gid & (RR - 1);
    int rest = gid >> 10;
    int c    = rest & (NCH - 1);
    int b    = rest >> 6;
    if (b >= BB) return;
    const float* base = zh + ((size_t)b * TT + (size_t)c * LCH) * (2 * RR);
    unsigned short* hb = ho + ((size_t)b * TT + (size_t)c * LCH) * RR + r;
    float h = hinit[gid];
    for (int i = 0; i < LCH; ++i) {
        float zl = base[(size_t)i * (2 * RR) + r];
        float hl = base[(size_t)i * (2 * RR) + RR + r];
        float z  = 1.f / (1.f + __expf(-zl));
        h = (1.f - z) * h + z * hl;
        hb[(size_t)i * RR] = f32_to_bf16_rne(h);
    }
}

// ---------------- host launcher ----------------
extern "C" void kernel_launch(void* const* d_in, const int* in_sizes, int n_in,
                              void* d_out, int out_size, void* d_ws, size_t ws_size,
                              hipStream_t stream) {
    const float* x      = (const float*)d_in[0];
    const float* w_zh   = (const float*)d_in[1];
    const float* w_out  = (const float*)d_in[2];
    const float* conv_w = (const float*)d_in[3];
    const float* conv_b = (const float*)d_in[4];
    float* out = (float*)d_out;

    char* ws = (char*)d_ws;
    const size_t OFF_WZH  = 0;
    const size_t OFF_WOUT = OFF_WZH  + (size_t)2 * RR * HH * 2;
    const size_t OFF_A1   = OFF_WOUT + (size_t)HH * RR * 2;
    const size_t OFF_ZH   = OFF_A1   + (size_t)MM * HH * 2;
    const size_t OFF_CA   = OFF_ZH   + (size_t)MM * 2 * RR * 4;
    const size_t OFF_CB   = OFF_CA   + (size_t)BB * NCH * RR * 4;
    const size_t OFF_HI   = OFF_CB   + (size_t)BB * NCH * RR * 4;

    unsigned short* wzh_bf  = (unsigned short*)(ws + OFF_WZH);
    unsigned short* wout_bf = (unsigned short*)(ws + OFF_WOUT);
    unsigned short* A1      = (unsigned short*)(ws + OFF_A1);
    unsigned short* ho      = A1;                 // buffer reuse after GEMM1
    float*          zh      = (float*)(ws + OFF_ZH);
    float*          cA      = (float*)(ws + OFF_CA);
    float*          cB      = (float*)(ws + OFF_CB);
    float*          hinit   = (float*)(ws + OFF_HI);

    k_cvt_w<<<(3u * 1024 * 1024 + 255) / 256, 256, 0, stream>>>(w_zh, w_out, wzh_bf, wout_bf);
    k_conv<<<(BB * (TT / 32) * HH) / 256, 256, 0, stream>>>(x, conv_w, conv_b, A1);
    k_gemm_bf16<<<dim3((2 * RR) / 128, MM / 128), 256, 0, stream>>>(A1, wzh_bf, zh, MM, 2 * RR, HH);
    k_scan1<<<(BB * NCH * RR) / 256, 256, 0, stream>>>(zh, cA, cB);
    k_scan2<<<(BB * RR + 255) / 256, 256, 0, stream>>>(cA, cB, hinit);
    k_scan3<<<(BB * NCH * RR) / 256, 256, 0, stream>>>(zh, hinit, ho);
    k_gemm_bf16<<<dim3(HH / 128, MM / 128), 256, 0, stream>>>(ho, wout_bf, out, MM, HH, RR);
}
